// _GRUODEEncoder_58085137711202
// MI455X (gfx1250) — matrix operand order repaired and checked by reading, not compiled, not run on hardware
//
#include <hip/hip_runtime.h>

#define B_ 1024
#define T_ 32
#define D_ 128
#define H_ 512
#define L_ 64
#define SH 520   /* LDS stride (bf16) for K=512 staging (bank-conflict padding) */
#define SX 136   /* LDS stride (bf16) for K=128 x staging */

typedef __attribute__((ext_vector_type(16))) __bf16          v16bf;
typedef __attribute__((ext_vector_type(8)))  float           v8f;
typedef __attribute__((ext_vector_type(8)))  unsigned short  us8;
typedef __attribute__((ext_vector_type(16))) unsigned short  us16;

__device__ __forceinline__ unsigned short f2bf(float f) {
  unsigned u = __builtin_bit_cast(unsigned, f);
  unsigned r = u + 0x7FFFu + ((u >> 16) & 1u);   // round-to-nearest-even
  return (unsigned short)(r >> 16);
}

__device__ __forceinline__ v8f vsplat(float v) {
  v8f r;
#pragma unroll
  for (int i = 0; i < 8; i++) r[i] = v;
  return r;
}

__device__ __forceinline__ float sigm1(float x) { return 1.f / (1.f + __expf(-x)); }
__device__ __forceinline__ float tanh1(float x) {
  x = fminf(15.f, fmaxf(-15.f, x));
  float e = __expf(2.f * x);
  return (e - 1.f) / (e + 1.f);
}

// ---------------------------------------------------------------------------
// Weight pack: W (N rows, K cols, row-major fp32) -> bf16 B-fragment order.
// Tile (nblk,kblk): element (lane,e) = W[nblk*16 + lane%16][kblk*32 + e + 16*(lane/16)]
// in contiguous k order, the same order gemm_acc uses for A; lane loads 32 contiguous B.
// ---------------------------------------------------------------------------
__global__ void pack_w_kernel(const float* __restrict__ W, unsigned short* __restrict__ dst,
                              int N, int K) {
  int idx = blockIdx.x * 256 + threadIdx.x;
  int total = N * K;
  if (idx >= total) return;
  int e    = idx & 15;
  int lane = (idx >> 4) & 31;
  int tile = idx >> 9;
  int KT   = K >> 5;
  int nblk = tile / KT;
  int kblk = tile - nblk * KT;
  int n = nblk * 16 + (lane & 15);
  int k = kblk * 32 + e + ((lane >> 4) << 4);
  dst[idx] = f2bf(W[(size_t)n * K + k]);
}

// ---------------------------------------------------------------------------
// Accumulating wave-level GEMM: o[NT] += A(16xK, LDS bf16) x B(Kx16*NT packed).
// kb loop kept ROLLED (unroll 1) to bound register pressure; nt loop unrolled
// for 4 independent WMMA accumulation chains per K-step.
// A-frag (contiguous K, matching packed B): lane L row = L%16,
//   K = kb*32 + e + 16*(L>=16).
// ---------------------------------------------------------------------------
template <int NT, int KT>
__device__ __forceinline__ void gemm_acc(v8f* o, const unsigned short* __restrict__ stg,
                                         int stride, const unsigned short* __restrict__ Wp,
                                         int nblk0, int lane) {
  const unsigned short* pa = stg + (lane & 15) * stride + ((lane >> 4) << 4);
  const unsigned short* pb = Wp + (size_t)nblk0 * KT * 512 + lane * 16;
#pragma unroll 1
  for (int kb = 0; kb < KT; kb++) {
    us8 lo = *(const us8*)pa;
    us8 hi = *(const us8*)(pa + 8);
    us16 au = __builtin_shufflevector(lo, hi, 0, 1, 2, 3, 4, 5, 6, 7,
                                      8, 9, 10, 11, 12, 13, 14, 15);
    v16bf a = __builtin_bit_cast(v16bf, au);
#pragma unroll
    for (int nt = 0; nt < NT; nt++) {
      v16bf b = __builtin_bit_cast(v16bf, *(const us16*)(pb + (size_t)nt * KT * 512));
      o[nt] = __builtin_amdgcn_wmma_f32_16x16x32_bf16(false, a, false, b, (short)0, o[nt],
                                                      false, false);
    }
    pa += 32;
    pb += 512;
  }
}

template <int NT>
__device__ __forceinline__ void bias_init(v8f* o, const float* __restrict__ bias, int nblk0,
                                          int lane) {
#pragma unroll
  for (int nt = 0; nt < NT; nt++) o[nt] = vsplat(bias[(nblk0 + nt) * 16 + (lane & 15)]);
}

template <int NT>
__device__ __forceinline__ void bias_init2(v8f* o, const float* __restrict__ ba,
                                           const float* __restrict__ bb, int nblk0, int lane) {
#pragma unroll
  for (int nt = 0; nt < NT; nt++) {
    int ix = (nblk0 + nt) * 16 + (lane & 15);
    o[nt] = vsplat(ba[ix] + bb[ix]);
  }
}

// C-frag layout: lane L, vgpr r -> (row = r + 8*(L/16), col = nt*16 + L%16).
__device__ __forceinline__ void write_chunk(unsigned short* stg, const v8f* c, int colBase,
                                            int lane) {
  int mB  = (lane >> 4) << 3;
  int col = colBase + (lane & 15);
#pragma unroll
  for (int nt = 0; nt < 4; nt++)
#pragma unroll
    for (int r = 0; r < 8; r++)
      stg[(mB + r) * SH + col + nt * 16] = f2bf(c[nt][r]);
}

// Store f2bf(a + coef*b) -- stages the RK4 intermediate state without holding
// y in registers.
__device__ __forceinline__ void write_chunk_fma(unsigned short* stg, const v8f* a, const v8f* b,
                                                float coef, int colBase, int lane) {
  int mB  = (lane >> 4) << 3;
  int col = colBase + (lane & 15);
#pragma unroll
  for (int nt = 0; nt < 4; nt++)
#pragma unroll
    for (int r = 0; r < 8; r++)
      stg[(mB + r) * SH + col + nt * 16] = f2bf(a[nt][r] + coef * b[nt][r]);
}

// ---------------------------------------------------------------------------
// Persistent GRU-ODE kernel: 64 blocks x 16 batch rows; 8 waves x 64 H-cols.
// ---------------------------------------------------------------------------
__global__ __launch_bounds__(256, 1) void gruode_kernel(
    const float* __restrict__ x, const float* __restrict__ t,
    const float* __restrict__ b1, const float* __restrict__ b2,
    const float* __restrict__ b_ih, const float* __restrict__ b_hh,
    const float* __restrict__ b_mu, const float* __restrict__ b_lv,
    const unsigned short* __restrict__ W1p, const unsigned short* __restrict__ W2p,
    const unsigned short* __restrict__ Wihp, const unsigned short* __restrict__ Whhp,
    const unsigned short* __restrict__ Wmup, const unsigned short* __restrict__ Wlvp,
    float* __restrict__ out) {
  __shared__ alignas(16) unsigned short stage[16 * SH];
  __shared__ alignas(16) unsigned short xstage[16 * SX];

  const int lane    = threadIdx.x & 31;
  const int wave    = threadIdx.x >> 5;
  const int rowBase = blockIdx.x * 16;
  const int colBase = wave * 64;
  const int nb0     = wave * 4;  // base 16-wide N-block within H for this wave

  v8f hc[4];
#pragma unroll
  for (int nt = 0; nt < 4; nt++) hc[nt] = vsplat(0.f);

#pragma unroll 1
  for (int step = 0; step < T_; step++) {
    float hs = (step > 0) ? (t[step] - t[step - 1]) * 0.25f : 0.f;

    // stage x_t (16 x 128) as bf16
    __syncthreads();
#pragma unroll 1
    for (int i = threadIdx.x; i < 16 * D_; i += 256) {
      int m = i >> 7, d = i & 127;
      xstage[m * SX + d] = f2bf(x[((size_t)(rowBase + m) * T_ + step) * D_ + d]);
    }
    __syncthreads();

    // ---- 4 RK4 substeps of dh/dt = W2 tanh(W1 h + b1) + b2 ----
#pragma unroll 1
    for (int sub = 0; sub < 4; sub++) {
      v8f hnext[4];
#pragma unroll
      for (int nt = 0; nt < 4; nt++) hnext[nt] = hc[nt];
      __syncthreads();
      write_chunk(stage, hc, colBase, lane);  // y1 = hc
      __syncthreads();
#pragma unroll 1
      for (int st = 0; st < 4; st++) {
        v8f k[4];
        bias_init<4>(k, b1, nb0, lane);
        gemm_acc<4, 16>(k, stage, SH, W1p, nb0, lane);   // t1 = y @ W1^T + b1
#pragma unroll
        for (int nt = 0; nt < 4; nt++)
#pragma unroll
          for (int i = 0; i < 8; i++) k[nt][i] = tanh1(k[nt][i]);
        __syncthreads();
        write_chunk(stage, k, colBase, lane);
        __syncthreads();
        bias_init<4>(k, b2, nb0, lane);
        gemm_acc<4, 16>(k, stage, SH, W2p, nb0, lane);   // k = tanh(...) @ W2^T + b2
        float wk = (hs * (1.f / 6.f)) * ((st == 0 || st == 3) ? 1.f : 2.f);
#pragma unroll
        for (int nt = 0; nt < 4; nt++)
#pragma unroll
          for (int i = 0; i < 8; i++) hnext[nt][i] += wk * k[nt][i];
        if (st < 3) {
          float c = (st == 2) ? hs : 0.5f * hs;
          __syncthreads();
          write_chunk_fma(stage, hc, k, c, colBase, lane);  // next y = hc + c*k
          __syncthreads();
        }
      }
#pragma unroll
      for (int nt = 0; nt < 4; nt++) hc[nt] = hnext[nt];
    }

    // ---- GRU cell: gates [r,z,n] at nblk offsets 0/32/64 within N=1536 ----
    __syncthreads();
    write_chunk(stage, hc, colBase, lane);
    __syncthreads();
    v8f rg[4], zg[4], tmp[4];
    // r = sigmoid(x@Wih_r + h@Whh_r + b_ih_r + b_hh_r)  (fused accumulator)
    bias_init2<4>(rg, b_ih, b_hh, 0 + nb0, lane);
    gemm_acc<4, 16>(rg, stage, SH, Whhp, 0 + nb0, lane);
    gemm_acc<4, 4>(rg, xstage, SX, Wihp, 0 + nb0, lane);
#pragma unroll
    for (int nt = 0; nt < 4; nt++)
#pragma unroll
      for (int i = 0; i < 8; i++) rg[nt][i] = sigm1(rg[nt][i]);
    // z = sigmoid(fused)
    bias_init2<4>(zg, b_ih, b_hh, 32 + nb0, lane);
    gemm_acc<4, 16>(zg, stage, SH, Whhp, 32 + nb0, lane);
    gemm_acc<4, 4>(zg, xstage, SX, Wihp, 32 + nb0, lane);
#pragma unroll
    for (int nt = 0; nt < 4; nt++)
#pragma unroll
      for (int i = 0; i < 8; i++) zg[nt][i] = sigm1(zg[nt][i]);
    // n = tanh(gi_n + r * gh_n):
    //   tmp = h@Whh_n + b_hh_n;  tmp = r*tmp + b_ih_n;  tmp += x@Wih_n
    bias_init<4>(tmp, b_hh, 64 + nb0, lane);
    gemm_acc<4, 16>(tmp, stage, SH, Whhp, 64 + nb0, lane);
#pragma unroll
    for (int nt = 0; nt < 4; nt++) {
      float bv = b_ih[(64 + nb0 + nt) * 16 + (lane & 15)];
#pragma unroll
      for (int i = 0; i < 8; i++) tmp[nt][i] = rg[nt][i] * tmp[nt][i] + bv;
    }
    gemm_acc<4, 4>(tmp, xstage, SX, Wihp, 64 + nb0, lane);
#pragma unroll
    for (int nt = 0; nt < 4; nt++)
#pragma unroll
      for (int i = 0; i < 8; i++) {
        float n = tanh1(tmp[nt][i]);
        hc[nt][i] = (1.f - zg[nt][i]) * n + zg[nt][i] * hc[nt][i];
      }
  }

  // ---- heads: waves 0-3 -> mu tile (wave&3); waves 4-7 -> clipped logvar ----
  __syncthreads();
  write_chunk(stage, hc, colBase, lane);
  __syncthreads();
  bool isLV = (wave >= 4);
  v8f o[1];
  bias_init<1>(o, isLV ? b_lv : b_mu, wave & 3, lane);
  gemm_acc<1, 16>(o, stage, SH, isLV ? Wlvp : Wmup, wave & 3, lane);
  if (isLV) {
#pragma unroll
    for (int i = 0; i < 8; i++) o[0][i] = fminf(5.f, fmaxf(-20.f, o[0][i]));
  }
  size_t base = isLV ? (size_t)B_ * L_ : 0;
  int col = (wave & 3) * 16 + (lane & 15);
  int mB = (lane >> 4) << 3;
#pragma unroll
  for (int r = 0; r < 8; r++)
    out[base + (size_t)(rowBase + mB + r) * L_ + col] = o[0][r];
}

// ---------------------------------------------------------------------------
extern "C" void kernel_launch(void* const* d_in, const int* in_sizes, int n_in,
                              void* d_out, int out_size, void* d_ws, size_t ws_size,
                              hipStream_t stream) {
  const float* x    = (const float*)d_in[0];
  const float* t    = (const float*)d_in[1];
  const float* W1   = (const float*)d_in[2];
  const float* b1   = (const float*)d_in[3];
  const float* W2   = (const float*)d_in[4];
  const float* b2   = (const float*)d_in[5];
  const float* W_ih = (const float*)d_in[6];
  const float* W_hh = (const float*)d_in[7];
  const float* b_ih = (const float*)d_in[8];
  const float* b_hh = (const float*)d_in[9];
  const float* W_mu = (const float*)d_in[10];
  const float* b_mu = (const float*)d_in[11];
  const float* W_lv = (const float*)d_in[12];
  const float* b_lv = (const float*)d_in[13];

  unsigned short* ws   = (unsigned short*)d_ws;
  unsigned short* W1p  = ws;                       // 512*512
  unsigned short* W2p  = W1p + 512 * 512;          // 512*512
  unsigned short* Wihp = W2p + 512 * 512;          // 1536*128
  unsigned short* Whhp = Wihp + 1536 * 128;        // 1536*512
  unsigned short* Wmup = Whhp + 1536 * 512;        // 64*512
  unsigned short* Wlvp = Wmup + 64 * 512;          // 64*512

  auto pack = [&](const float* W, unsigned short* dst, int N, int K) {
    int total = N * K;
    pack_w_kernel<<<(total + 255) / 256, 256, 0, stream>>>(W, dst, N, K);
  };
  pack(W1, W1p, 512, 512);
  pack(W2, W2p, 512, 512);
  pack(W_ih, Wihp, 1536, 128);
  pack(W_hh, Whhp, 1536, 512);
  pack(W_mu, Wmup, 64, 512);
  pack(W_lv, Wlvp, 64, 512);

  gruode_kernel<<<B_ / 16, 256, 0, stream>>>(x, t, b1, b2, b_ih, b_hh, b_mu, b_lv,
                                             W1p, W2p, Wihp, Whhp, Wmup, Wlvp,
                                             (float*)d_out);
}
